// HessianDiagonal_16501264351434
// MI455X (gfx1250) — matrix operand order repaired and checked by reading, not compiled, not run on hardware
//
#include <hip/hip_runtime.h>
#include <hip/hip_bf16.h>

// ---------------------------------------------------------------------------
// Gated equivariant blocks (PaiNN-style output head), fused for gfx1250.
//   N_ATOMS=262144, N_IN=128, H1=64
// One wave32 = one 16-atom tile. All matmuls via v_wmma_f32_16x16x32_f16.
// ---------------------------------------------------------------------------

#define N_ATOMS 262144
#define WAVES   4          // waves per block
#define BLOCK   (WAVES * 32)
#define ATOMS_PER_BLOCK (WAVES * 16)

typedef _Float16 v16h __attribute__((ext_vector_type(16)));
typedef _Float16 v8h  __attribute__((ext_vector_type(8)));
typedef _Float16 v4h  __attribute__((ext_vector_type(4)));
typedef float    v8f  __attribute__((ext_vector_type(8)));
typedef float    v4f  __attribute__((ext_vector_type(4)));

// ws layout (in halves): Wv1 frags [0,16384) ; W1a [16384,28672) ; W1b [28672,36864)
#define WSH_WV1 0
#define WSH_W1A 16384
#define WSH_W1B 28672
#define WSH_TOTAL 36864
// same bases in v16h (16-half) units:
#define WSU_WV1 0
#define WSU_W1A 1024
#define WSU_W1B 1792

__device__ __forceinline__ float silu_f(float x) {
    return x / (1.0f + __expf(-x));
}

__device__ __forceinline__ v8f wmma16(v16h a, v16h b, v8f c) {
    return __builtin_amdgcn_wmma_f32_16x16x32_f16(
        /*neg_a=*/false, a, /*neg_b=*/false, b,
        /*c_mod=*/(short)0, c, /*reuse_a=*/false, /*reuse_b=*/false);
}

// A-fragment (16x32 f16) from row-major f32 global memory.
// 16-bit A layout: lanes 0-15 hold row M=lane, K = {kb..kb+7, kb+16..kb+23};
// lanes 16-31 hold row M=lane-16, K = {kb+8..kb+15, kb+24..kb+31}.
__device__ __forceinline__ v16h load_a_frag_g(const float* __restrict__ rowbase,
                                              int rs, int kb, int m, int hi) {
    const float* p = rowbase + (long)m * rs + kb + hi * 8;
    v4f x0 = *(const v4f*)(p);
    v4f x1 = *(const v4f*)(p + 4);
    v4f y0 = *(const v4f*)(p + 16);
    v4f y1 = *(const v4f*)(p + 20);
    v16h r;
#pragma unroll
    for (int i = 0; i < 4; ++i) {
        r[i]      = (_Float16)x0[i];
        r[4 + i]  = (_Float16)x1[i];
        r[8 + i]  = (_Float16)y0[i];
        r[12 + i] = (_Float16)y1[i];
    }
    return r;
}

// A-fragment from f16 LDS (row-major, stride rs halves).
__device__ __forceinline__ v16h load_a_frag_lds(const _Float16* base,
                                                int rs, int kb, int m, int hi) {
    const _Float16* p = base + m * rs + kb + hi * 8;
    v8h lo = *(const v8h*)(p);        // ds_load_b128
    v8h hv = *(const v8h*)(p + 16);   // ds_load_b128
    return __builtin_shufflevector(lo, hv, 0,1,2,3,4,5,6,7,8,9,10,11,12,13,14,15);
}

// ---------------------------------------------------------------------------
// Kernel 0: stage Wv1/W1a/W1b (f32) into d_ws as f16 in B-fragment lane order.
// B layout (32x16 f16): lane = 16*hi + n holds column N=n, K = kb + 8*hi + (j & 7) + 16*(j >> 3)
// for j=0..15 (AMD table order). Fragment f at half-index base + f*512 + lane*16 + j.
// ---------------------------------------------------------------------------
__global__ void stage_weights_kernel(const float* __restrict__ Wv1,
                                     const float* __restrict__ W1a,
                                     const float* __restrict__ W1b,
                                     _Float16* __restrict__ ws) {
    int t = blockIdx.x * blockDim.x + threadIdx.x;
    if (t >= WSH_TOTAL) return;
    const float* W;
    int base, NT, Ntot;
    if (t < WSH_W1A)      { W = Wv1; base = WSH_WV1; NT = 8; Ntot = 128; }
    else if (t < WSH_W1B) { W = W1a; base = WSH_W1A; NT = 4; Ntot = 64;  }
    else                  { W = W1b; base = WSH_W1B; NT = 8; Ntot = 128; }
    int e    = t - base;
    int f    = e >> 9;
    int lane = (e >> 4) & 31;
    int j    = e & 15;
    int kt = f / NT, nt = f % NT;
    int K = kt * 32 + ((lane & 16) ? 8 : 0) + (j & 7) + ((j & 8) ? 16 : 0);
    int N = nt * 16 + (lane & 15);
    ws[t] = (_Float16)W[K * Ntot + N];
}

// ---------------------------------------------------------------------------
// Main fused kernel.
// Per-wave LDS (halves): buf0[3072] = ctx (16x192), later reused as
// s2(16x64)@0 + gate(16x64)@1024 ; vw[3072] = vW (16 x 3 x 64) ; hb[1024] = h.
// ---------------------------------------------------------------------------
__global__ __launch_bounds__(BLOCK) void gated_equiv_kernel(
    const float* __restrict__ l0,   // [N,128]
    const float* __restrict__ l1,   // [N,3,128]
    const _Float16* __restrict__ ws,
    const float* __restrict__ b1a,  // [64]
    const float* __restrict__ b1b,  // [128]
    const float* __restrict__ Wv2,  // [64,2]
    const float* __restrict__ W2a,  // [65]
    const float* __restrict__ b2a,  // [1]
    const float* __restrict__ W2b,  // [2]
    const float* __restrict__ b2b,  // [2]
    float* __restrict__ out)        // [N*3]
{
    __shared__ __align__(16) _Float16 smem[WAVES * 7168];

    const int tid  = threadIdx.x;
    const int lane = tid & 31;
    const int wv   = tid >> 5;
    const int l15  = lane & 15;
    const int hi   = (lane & 16) ? 1 : 0;

    _Float16* buf0 = smem + wv * 7168;   // ctx / (s2, gate)
    _Float16* vwb  = buf0 + 3072;        // vW
    _Float16* hb   = buf0 + 6144;        // h

    const long gw   = (long)blockIdx.x * WAVES + wv;
    const long base = gw * 16;

    const v16h* wsv = (const v16h*)ws;

    // ---- stage s = l0 tile into ctx cols [0,128) as f16 (coop over 32 lanes)
#pragma unroll 4
    for (int m = 0; m < 16; ++m) {
        v4f s4 = *(const v4f*)(l0 + (base + m) * 128 + lane * 4);
        v4h h4;
#pragma unroll
        for (int i = 0; i < 4; ++i) h4[i] = (_Float16)s4[i];
        *(v4h*)(buf0 + m * 192 + lane * 4) = h4;
    }

    // ---- load A fragments of l1 (3 components x 4 K-chunks), f32 -> f16
    v16h A[3][4];
#pragma unroll
    for (int a = 0; a < 3; ++a) {
        const float* pa = l1 + base * 384 + a * 128;
#pragma unroll
        for (int kt = 0; kt < 4; ++kt)
            A[a][kt] = load_a_frag_g(pa, 384, kt * 32, l15, hi);
    }

    // ---- vmix = v @ Wv1 : nt 0..3 -> vV (norm^2 accum), nt 4..7 -> vW (LDS)
    v8f vsq[4];
#pragma unroll
    for (int i = 0; i < 4; ++i) vsq[i] = (v8f)(0.0f);

#pragma unroll
    for (int nt = 0; nt < 8; ++nt) {
        v16h B0 = wsv[WSU_WV1 + (0 * 8 + nt) * 32 + lane];
        v16h B1 = wsv[WSU_WV1 + (1 * 8 + nt) * 32 + lane];
        v16h B2 = wsv[WSU_WV1 + (2 * 8 + nt) * 32 + lane];
        v16h B3 = wsv[WSU_WV1 + (3 * 8 + nt) * 32 + lane];
#pragma unroll
        for (int a = 0; a < 3; ++a) {
            v8f c = (v8f)(0.0f);
            c = wmma16(A[a][0], B0, c);
            c = wmma16(A[a][1], B1, c);
            c = wmma16(A[a][2], B2, c);
            c = wmma16(A[a][3], B3, c);
            if (nt < 4) {
                vsq[nt] += c * c;
            } else {
                int col = (nt - 4) * 16 + l15;
#pragma unroll
                for (int r = 0; r < 8; ++r) {
                    int m = r + hi * 8;
                    vwb[(m * 3 + a) * 64 + col] = (_Float16)c[r];
                }
            }
        }
    }

    // ---- vVn = sqrt(sum_a vV^2) -> ctx cols [128,192)
#pragma unroll
    for (int nt = 0; nt < 4; ++nt) {
#pragma unroll
        for (int r = 0; r < 8; ++r) {
            int m = r + hi * 8;
            buf0[m * 192 + 128 + nt * 16 + l15] = (_Float16)__fsqrt_rn(vsq[nt][r]);
        }
    }

    // ---- h = silu(ctx @ W1a + b1a)   (16x192 @ 192x64)
    v16h AC[6];
#pragma unroll
    for (int kt = 0; kt < 6; ++kt)
        AC[kt] = load_a_frag_lds(buf0, 192, kt * 32, l15, hi);

#pragma unroll
    for (int nt2 = 0; nt2 < 4; ++nt2) {
        v8f c = (v8f)(0.0f);
#pragma unroll
        for (int kt = 0; kt < 6; ++kt)
            c = wmma16(AC[kt], wsv[WSU_W1A + (kt * 4 + nt2) * 32 + lane], c);
        float bv = b1a[nt2 * 16 + l15];
        int col = nt2 * 16 + l15;
#pragma unroll
        for (int r = 0; r < 8; ++r) {
            int m = r + hi * 8;
            hb[m * 64 + col] = (_Float16)silu_f(c[r] + bv);
        }
    }

    // ---- y = h @ W1b + b1b : nt3 0..3 -> s2 = silu(y), nt3 4..7 -> gate
    v16h AH[2];
#pragma unroll
    for (int kt = 0; kt < 2; ++kt)
        AH[kt] = load_a_frag_lds(hb, 64, kt * 32, l15, hi);

#pragma unroll
    for (int nt3 = 0; nt3 < 8; ++nt3) {
        v8f c = (v8f)(0.0f);
#pragma unroll
        for (int kt = 0; kt < 2; ++kt)
            c = wmma16(AH[kt], wsv[WSU_W1B + (kt * 8 + nt3) * 32 + lane], c);
        float bv = b1b[nt3 * 16 + l15];
#pragma unroll
        for (int r = 0; r < 8; ++r) {
            int m = r + hi * 8;
            float y = c[r] + bv;
            if (nt3 < 4) {  // s2 (overwrites dead ctx region)
                buf0[m * 64 + nt3 * 16 + l15] = (_Float16)silu_f(y);
            } else {        // gate
                buf0[1024 + m * 64 + (nt3 - 4) * 16 + l15] = (_Float16)y;
            }
        }
    }

    // ---- block 2 tail: one lane per atom (lanes 0..15)
    if (lane < 16) {
        const int m = lane;
        float vW2a[3];
        float vv2 = 0.0f;
#pragma unroll
        for (int a = 0; a < 3; ++a) {
            float sv = 0.0f, sw = 0.0f;
#pragma unroll 8
            for (int j = 0; j < 64; ++j) {
                float g  = (float)buf0[1024 + m * 64 + j];
                float w  = (float)vwb[(m * 3 + a) * 64 + j];
                float vo = g * w;                 // v_out1 element
                sv = fmaf(vo, Wv2[2 * j + 0], sv); // -> vV2
                sw = fmaf(vo, Wv2[2 * j + 1], sw); // -> vW2
            }
            vv2 += sv * sv;
            vW2a[a] = sw;
        }
        float vvn2 = __fsqrt_rn(vv2);
        float acc = b2a[0] + vvn2 * W2a[64];
#pragma unroll 8
        for (int k = 0; k < 64; ++k)
            acc = fmaf((float)buf0[m * 64 + k], W2a[k], acc);
        float h2 = silu_f(acc);
        float g2 = fmaf(h2, W2b[1], b2b[1]);
#pragma unroll
        for (int a = 0; a < 3; ++a)
            out[(base + m) * 3 + a] = 1000.0f * g2 * vW2a[a];
    }
}

// ---------------------------------------------------------------------------
extern "C" void kernel_launch(void* const* d_in, const int* in_sizes, int n_in,
                              void* d_out, int out_size, void* d_ws, size_t ws_size,
                              hipStream_t stream) {
    const float* l0  = (const float*)d_in[0];
    const float* l1  = (const float*)d_in[1];
    // d_in[2] = idx_m (int64) — unused by the reference output
    const float* Wv1 = (const float*)d_in[3];
    const float* W1a = (const float*)d_in[4];
    const float* b1a = (const float*)d_in[5];
    const float* W1b = (const float*)d_in[6];
    const float* b1b = (const float*)d_in[7];
    const float* Wv2 = (const float*)d_in[8];
    const float* W2a = (const float*)d_in[9];
    const float* b2a = (const float*)d_in[10];
    const float* W2b = (const float*)d_in[11];
    const float* b2b = (const float*)d_in[12];
    float* out = (float*)d_out;
    _Float16* ws = (_Float16*)d_ws;

    // Stage weights into f16 B-fragment order (73728 B of d_ws).
    stage_weights_kernel<<<(WSH_TOTAL + 255) / 256, 256, 0, stream>>>(Wv1, W1a, W1b, ws);

    const int grid = N_ATOMS / ATOMS_PER_BLOCK;  // 4096
    gated_equiv_kernel<<<grid, BLOCK, 0, stream>>>(
        l0, l1, ws, b1a, b1b, Wv2, W2a, b2a, W2b, b2b, out);
}
